// LayoutLMv2Encoder_6296422056372
// MI455X (gfx1250) — matrix operand order repaired and checked by reading, not compiled, not run on hardware
//
#include <hip/hip_runtime.h>
#include <math.h>

// LayoutLMv2 encoder forward on gfx1250 (MI455X), bf16 WMMA everywhere.
// Dims: L=12, D=768, H=12, DH=64, FF=3072, B=8, N=512.  M = B*N = 4096.

typedef __attribute__((ext_vector_type(16))) __bf16 v16bf;
typedef __attribute__((ext_vector_type(8)))  float  v8f;
typedef __attribute__((ext_vector_type(4)))  unsigned int u32x4;
typedef __attribute__((ext_vector_type(2)))  unsigned int u32x2;
typedef __attribute__((ext_vector_type(4)))  float  f32x4;
typedef __attribute__((ext_vector_type(4)))  int    i32x4;

union Frag {
    v16bf v;
    u32x4 q[2];
    unsigned int u[8];
    unsigned short s[16];
};

__device__ __forceinline__ unsigned short f2bf(float f) {
    union { float f; unsigned int u; } x; x.f = f;
    unsigned int u = x.u;
    u += 0x7FFFu + ((u >> 16) & 1u);        // round-to-nearest-even
    return (unsigned short)(u >> 16);
}

__device__ __forceinline__ unsigned int pack2bf(float lo, float hi) {
    return (unsigned int)f2bf(lo) | ((unsigned int)f2bf(hi) << 16);
}

__device__ __forceinline__ v8f vzero8() {
    v8f z = {0.f, 0.f, 0.f, 0.f, 0.f, 0.f, 0.f, 0.f};
    return z;
}

// ---------------------------------------------------------------------------
// Relative-position bucket (faithful to reference helper), packed per (b,i,j):
//   bits [4:0]=1D bucket, [10:5]=x bucket, [16:11]=y bucket, [17]=mask
// ---------------------------------------------------------------------------
__device__ __forceinline__ int rel_bucket(int rel, int nbuckets, float maxdist) {
    int nb = nbuckets >> 1;
    int ret = rel > 0 ? nb : 0;
    int n = rel < 0 ? -rel : rel;
    int max_exact = nb >> 1;
    int nc = n < 1 ? 1 : n;
    float lv = __logf((float)nc / (float)max_exact) /
               __logf(maxdist / (float)max_exact) * (float)(nb - max_exact);
    int vl = max_exact + (int)lv;            // truncation == floor (lv >= 0 when used)
    vl = vl < nb - 1 ? vl : nb - 1;
    return ret + (n < max_exact ? n : vl);
}

__global__ __launch_bounds__(256) void build_idx(
    const int* __restrict__ pos, const int* __restrict__ bbox,
    const int* __restrict__ amask, int* __restrict__ pidx)
{
    int t = blockIdx.x * 256 + threadIdx.x;          // B*N*N = 2,097,152
    int b = t >> 18;
    int r = t & 262143;
    int i = r >> 9, j = r & 511;
    int pi = pos[b * 512 + i], pj = pos[b * 512 + j];
    int b1 = rel_bucket(pj - pi, 32, 128.f);
    const int* bi = bbox + ((size_t)b * 512 + i) * 4;
    const int* bj = bbox + ((size_t)b * 512 + j) * 4;
    int bx = rel_bucket(bj[0] - bi[0], 64, 256.f);
    int by = rel_bucket(bj[3] - bi[3], 64, 256.f);
    int m = amask[b * 512 + j] != 0 ? 1 : 0;
    pidx[t] = b1 | (bx << 5) | (by << 11) | (m << 17);
}

// ---------------------------------------------------------------------------
// bf16 WMMA GEMM: C[4096 x Nc] = A[4096 x K] * W[K x Nc] (+ epilogue)
// Block tile 128x128, 8 waves (each 32x64), K staged in chunks of 32 via LDS.
// A-frag (16x32): lane m=l%16; e<8 -> k=8*hb+e, e>=8 -> k=16+8*hb+(e-8)
// B-frag (32x16): lane n=l%16; e<8 -> k=8*hb+e, e>=8 -> k=16+8*hb+(e-8)
// C-frag (16x16): lane n=l%16, m = g + 8*hb
// EPI: 0 = QKV split ((q+qbias)*0.125 -> q[B,H,N,DH]; k; v+vbias -> vT[B,H,DH,N])
//      1 = + bias[col] + resid[row,col] -> fp32
//      2 = gelu_exact(x + bias[col])    -> fp32
// ---------------------------------------------------------------------------
template <int EPI>
__global__ __launch_bounds__(256) void gemm_bf16_wmma(
    const float* __restrict__ A, const float* __restrict__ W,
    int K, int Nc,
    const float* __restrict__ bias, const float* __restrict__ resid,
    float* __restrict__ outF,
    const float* __restrict__ qbias, const float* __restrict__ vbias,
    unsigned short* __restrict__ qout, unsigned short* __restrict__ kout,
    unsigned short* __restrict__ vout)
{
    __shared__ __align__(16) unsigned short As[128 * 40];   // [row][k], pad 8
    __shared__ __align__(16) unsigned short Bs[128 * 40];   // [col][k], pad 8
    const int tid = threadIdx.x;
    const int rowBase = blockIdx.y * 128;
    const int colBase = blockIdx.x * 128;
    const int wave = tid >> 5, lane = tid & 31;
    const int nl = lane & 15, hb = lane >> 4;
    const int wm = wave >> 1, wn = wave & 1;

    v8f acc[2][4];
#pragma unroll
    for (int tm = 0; tm < 2; ++tm)
#pragma unroll
        for (int tn = 0; tn < 4; ++tn) acc[tm][tn] = vzero8();

    for (int k0 = 0; k0 < K; k0 += 32) {
        // A tile 128x32 fp32 -> bf16 LDS: 4 x b128 global loads + 4 x b64 LDS stores
#pragma unroll
        for (int it = 0; it < 4; ++it) {
            int idx4 = it * 256 + tid;                       // 1024 float4s
            int r = idx4 >> 3;                               // 8 float4 per row
            int kc = (idx4 & 7) << 2;
            f32x4 a = *(const f32x4*)(A + (size_t)(rowBase + r) * K + k0 + kc);
            u32x2 p;
            p[0] = pack2bf(a[0], a[1]);
            p[1] = pack2bf(a[2], a[3]);
            *(u32x2*)&As[r * 40 + kc] = p;                   // 8B aligned
        }
        // W tile 32x128 fp32 -> bf16 LDS transposed [n][k]
#pragma unroll
        for (int it = 0; it < 4; ++it) {
            int idx4 = it * 256 + tid;
            int kk = idx4 >> 5;                              // 32 float4 per k-row
            int n0 = (idx4 & 31) << 2;
            f32x4 wv = *(const f32x4*)(W + (size_t)(k0 + kk) * Nc + colBase + n0);
            Bs[(n0 + 0) * 40 + kk] = f2bf(wv[0]);
            Bs[(n0 + 1) * 40 + kk] = f2bf(wv[1]);
            Bs[(n0 + 2) * 40 + kk] = f2bf(wv[2]);
            Bs[(n0 + 3) * 40 + kk] = f2bf(wv[3]);
        }
        __syncthreads();

        Frag af[2], bfr[4];
#pragma unroll
        for (int tm = 0; tm < 2; ++tm) {
            const unsigned short* ap = &As[(wm * 32 + tm * 16 + nl) * 40 + 8 * hb];
            af[tm].q[0] = *(const u32x4*)ap;
            af[tm].q[1] = *(const u32x4*)(ap + 16);
        }
#pragma unroll
        for (int tn = 0; tn < 4; ++tn) {
            const unsigned short* bp = &Bs[(wn * 64 + tn * 16 + nl) * 40 + 8 * hb];
            bfr[tn].q[0] = *(const u32x4*)bp;
            bfr[tn].q[1] = *(const u32x4*)(bp + 16);
        }
#pragma unroll
        for (int tm = 0; tm < 2; ++tm)
#pragma unroll
            for (int tn = 0; tn < 4; ++tn)
                acc[tm][tn] = __builtin_amdgcn_wmma_f32_16x16x32_bf16(
                    false, af[tm].v, false, bfr[tn].v, (short)0, acc[tm][tn],
                    false, false);
        __syncthreads();
    }

#pragma unroll
    for (int tm = 0; tm < 2; ++tm)
#pragma unroll
        for (int tn = 0; tn < 4; ++tn)
#pragma unroll
            for (int g = 0; g < 8; ++g) {
                int row = rowBase + wm * 32 + tm * 16 + g + 8 * hb;
                int col = colBase + wn * 64 + tn * 16 + nl;
                float val = acc[tm][tn][g];
                if (EPI == 0) {
                    int part = col / 768, d = col % 768;
                    int hh = d >> 6, dh = d & 63;
                    int bb = row >> 9, n = row & 511;
                    if (part == 0) {
                        float v = (val + qbias[d]) * 0.125f;   // 1/sqrt(64)
                        qout[(((size_t)bb * 12 + hh) * 512 + n) * 64 + dh] = f2bf(v);
                    } else if (part == 1) {
                        kout[(((size_t)bb * 12 + hh) * 512 + n) * 64 + dh] = f2bf(val);
                    } else {
                        float v = val + vbias[d];
                        vout[(((size_t)bb * 12 + hh) * 64 + dh) * 512 + n] = f2bf(v);
                    }
                } else if (EPI == 1) {
                    outF[(size_t)row * Nc + col] =
                        val + bias[col] + resid[(size_t)row * Nc + col];
                } else {
                    float x = val + bias[col];
                    outF[(size_t)row * Nc + col] =
                        0.5f * x * (1.0f + erff(x * 0.7071067811865476f));
                }
            }
}

// ---------------------------------------------------------------------------
// Flash attention, one wave = 16 queries.  Computes S^T = K_tile @ Q^T so the
// softmax reduction over keys is in-register; ctx^T = V^T @ P^T.
// q,k: bf16 [B,H,N,DH]; vT: bf16 [B,H,DH,N]; ctx: fp32 [B,N,D].
// ---------------------------------------------------------------------------
__global__ __launch_bounds__(128) void attn_flash(
    const unsigned short* __restrict__ q, const unsigned short* __restrict__ k,
    const unsigned short* __restrict__ vT, const int* __restrict__ pidx,
    const float* __restrict__ rel1, const float* __restrict__ relx,
    const float* __restrict__ rely, float* __restrict__ ctx)
{
    __shared__ float t1[32], tx[64], ty[64];
    const int b = blockIdx.z, h = blockIdx.y;
    const int tid = threadIdx.x;
    if (tid < 32) t1[tid] = rel1[h * 32 + tid];
    if (tid < 64) tx[tid] = relx[h * 64 + tid];
    else          ty[tid - 64] = rely[h * 64 + tid - 64];
    __syncthreads();

    const int wave = tid >> 5, lane = tid & 31, nl = lane & 15, hb = lane >> 4;
    const int i = blockIdx.x * 64 + wave * 16 + nl;       // query index
    const int bh = b * 12 + h;

    // Q as B-fragments (dh chunks of 32): lane col = i, dh = c*32+8*hb+e (e<8), c*32+16+8*hb+(e-8) (e>=8)
    const unsigned short* qrow = q + ((size_t)bh * 512 + i) * 64;
    Frag qf[2];
#pragma unroll
    for (int c = 0; c < 2; ++c) {
        const unsigned short* p = qrow + c * 32 + 8 * hb;
        qf[c].q[0] = *(const u32x4*)p;
        qf[c].q[1] = *(const u32x4*)(p + 16);
    }

    v8f o[4];
#pragma unroll
    for (int c = 0; c < 4; ++c) o[c] = vzero8();
    float mrun = -3.0e38f, lrun = 0.0f;
    const int* idxrow = pidx + ((size_t)b * 512 + i) * 512;

    for (int jc = 0; jc < 512; jc += 32) {
        v8f s0 = vzero8(), s1 = vzero8();
        {   // K rows as A-fragments; contraction over dh (two 32-chunks)
            const unsigned short* kr0 = k + ((size_t)bh * 512 + jc + nl) * 64;
            Frag a;
            a.q[0] = *(const u32x4*)(kr0 + 8 * hb);
            a.q[1] = *(const u32x4*)(kr0 + 16 + 8 * hb);
            s0 = __builtin_amdgcn_wmma_f32_16x16x32_bf16(false, a.v, false, qf[0].v, (short)0, s0, false, false);
            a.q[0] = *(const u32x4*)(kr0 + 32 + 8 * hb);
            a.q[1] = *(const u32x4*)(kr0 + 48 + 8 * hb);
            s0 = __builtin_amdgcn_wmma_f32_16x16x32_bf16(false, a.v, false, qf[1].v, (short)0, s0, false, false);
            const unsigned short* kr1 = kr0 + 16 * 64;
            a.q[0] = *(const u32x4*)(kr1 + 8 * hb);
            a.q[1] = *(const u32x4*)(kr1 + 16 + 8 * hb);
            s1 = __builtin_amdgcn_wmma_f32_16x16x32_bf16(false, a.v, false, qf[0].v, (short)0, s1, false, false);
            a.q[0] = *(const u32x4*)(kr1 + 32 + 8 * hb);
            a.q[1] = *(const u32x4*)(kr1 + 48 + 8 * hb);
            s1 = __builtin_amdgcn_wmma_f32_16x16x32_bf16(false, a.v, false, qf[1].v, (short)0, s1, false, false);
        }
        // packed bucket indices: lane's 8-element groups are contiguous ints
        const int* ip0 = idxrow + jc + 8 * hb;
        i32x4 ia = *(const i32x4*)ip0;
        i32x4 ibv = *(const i32x4*)(ip0 + 4);
        const int* ip1 = idxrow + jc + 16 + 8 * hb;
        i32x4 ic = *(const i32x4*)ip1;
        i32x4 idv = *(const i32x4*)(ip1 + 4);
        int id0[8], id1[8];
#pragma unroll
        for (int g = 0; g < 4; ++g) {
            id0[g] = ia[g]; id0[g + 4] = ibv[g];
            id1[g] = ic[g]; id1[g + 4] = idv[g];
        }
        // bias + mask; lane holds j = jc + g + 8*hb (s0) and +16 (s1)
        float p0[8], p1[8];
        float cmax = -3.0e38f;
#pragma unroll
        for (int g = 0; g < 8; ++g) {
            float bs0 = t1[id0[g] & 31] + tx[(id0[g] >> 5) & 63] + ty[(id0[g] >> 11) & 63];
            float sv0 = (id0[g] >> 17) ? 1e-8f : s0[g] + bs0;
            float bs1 = t1[id1[g] & 31] + tx[(id1[g] >> 5) & 63] + ty[(id1[g] >> 11) & 63];
            float sv1 = (id1[g] >> 17) ? 1e-8f : s1[g] + bs1;
            p0[g] = sv0; p1[g] = sv1;
            cmax = fmaxf(cmax, fmaxf(sv0, sv1));
        }
        cmax = fmaxf(cmax, __shfl_xor(cmax, 16, 32));     // merge half-lane j-partials
        float mnew = fmaxf(mrun, cmax);
        float scale = __expf(mrun - mnew);
        float lsum = 0.0f;
#pragma unroll
        for (int g = 0; g < 8; ++g) {
            p0[g] = __expf(p0[g] - mnew);
            p1[g] = __expf(p1[g] - mnew);
            lsum += p0[g] + p1[g];
        }
        lsum += __shfl_xor(lsum, 16, 32);
        lrun = lrun * scale + lsum;
        mrun = mnew;
#pragma unroll
        for (int c = 0; c < 4; ++c) o[c] *= scale;

        // Assemble P^T B-fragment: lane col = i, k(e) = j_local = e + 16*hb.
        // hb=0 needs j 0..15 (own s0 + partner s0); hb=1 needs 16..31.
        unsigned int a0p[4], a1p[4];
#pragma unroll
        for (int gg = 0; gg < 4; ++gg) {
            a0p[gg] = pack2bf(p0[2 * gg], p0[2 * gg + 1]);
            a1p[gg] = pack2bf(p1[2 * gg], p1[2 * gg + 1]);
        }
        Frag pf;
#pragma unroll
        for (int gg = 0; gg < 4; ++gg) {
            unsigned int o0 = __shfl_xor(a0p[gg], 16, 32);
            unsigned int o1 = __shfl_xor(a1p[gg], 16, 32);
            pf.u[gg]     = hb ? o1      : a0p[gg];
            pf.u[4 + gg] = hb ? a1p[gg] : o0;
        }
        // ctx^T += V^T (A-frags, m = dh) @ P^T
#pragma unroll
        for (int c = 0; c < 4; ++c) {
            const unsigned short* vr = vT + ((size_t)bh * 64 + c * 16 + nl) * 512 + jc;
            Frag vf;
            vf.q[0] = *(const u32x4*)(vr + 16 * hb);
            vf.q[1] = *(const u32x4*)(vr + 16 * hb + 8);
            o[c] = __builtin_amdgcn_wmma_f32_16x16x32_bf16(false, vf.v, false, pf.v, (short)0, o[c], false, false);
        }
    }

    float inv = 1.0f / lrun;
    float* cr = ctx + ((size_t)b * 512 + i) * 768 + h * 64;
#pragma unroll
    for (int c = 0; c < 4; ++c)
#pragma unroll
        for (int g = 0; g < 8; ++g)
            cr[c * 16 + 8 * hb + g] = o[c][g] * inv;      // dh = c*16 + g + 8*hb
}

// ---------------------------------------------------------------------------
// LayerNorm over D=768 (population variance, eps=1e-12), one block per row.
// ---------------------------------------------------------------------------
__global__ __launch_bounds__(256) void layernorm_768(
    const float* __restrict__ x, const float* __restrict__ g,
    const float* __restrict__ bta, float* __restrict__ y)
{
    __shared__ float rs[256], rq[256];
    const int row = blockIdx.x, tid = threadIdx.x;
    const float* xr = x + (size_t)row * 768;
    float s = 0.f, ss = 0.f;
#pragma unroll
    for (int d = tid; d < 768; d += 256) { float v = xr[d]; s += v; ss += v * v; }
    rs[tid] = s; rq[tid] = ss;
    __syncthreads();
    for (int off = 128; off > 0; off >>= 1) {
        if (tid < off) { rs[tid] += rs[tid + off]; rq[tid] += rq[tid + off]; }
        __syncthreads();
    }
    float mean = rs[0] * (1.0f / 768.0f);
    float var  = rq[0] * (1.0f / 768.0f) - mean * mean;
    float inv  = rsqrtf(var + 1e-12f);
    float* yr = y + (size_t)row * 768;
    for (int d = tid; d < 768; d += 256)
        yr[d] = (xr[d] - mean) * inv * g[d] + bta[d];
}

// ---------------------------------------------------------------------------
extern "C" void kernel_launch(void* const* d_in, const int* in_sizes, int n_in,
                              void* d_out, int out_size, void* d_ws, size_t ws_size,
                              hipStream_t stream) {
    (void)in_sizes; (void)n_in; (void)out_size; (void)ws_size;
    const float* hidden     = (const float*)d_in[0];
    const float* qkv_w      = (const float*)d_in[1];
    const float* q_bias     = (const float*)d_in[2];
    const float* v_bias     = (const float*)d_in[3];
    const float* attn_out_w = (const float*)d_in[4];
    const float* attn_out_b = (const float*)d_in[5];
    const float* ln1_g      = (const float*)d_in[6];
    const float* ln1_b      = (const float*)d_in[7];
    const float* inter_w    = (const float*)d_in[8];
    const float* inter_b    = (const float*)d_in[9];
    const float* out_w      = (const float*)d_in[10];
    const float* out_b      = (const float*)d_in[11];
    const float* ln2_g      = (const float*)d_in[12];
    const float* ln2_b      = (const float*)d_in[13];
    const float* rel_pos_w  = (const float*)d_in[14];
    const float* rel_x_w    = (const float*)d_in[15];
    const float* rel_y_w    = (const float*)d_in[16];
    const int* position_ids = (const int*)d_in[17];
    const int* bbox         = (const int*)d_in[18];
    const int* amask        = (const int*)d_in[19];

    float* h = (float*)d_out;                       // running hidden [4096,768]

    char* w = (char*)d_ws;
    size_t off = 0;
    auto alloc = [&](size_t bytes) -> void* {
        void* p = w + off;
        off += (bytes + 255) & ~(size_t)255;
        return p;
    };
    int*            pidx  = (int*)alloc(8ull * 512 * 512 * 4);            // 8.4 MB
    unsigned short* qb    = (unsigned short*)alloc(8ull * 12 * 512 * 64 * 2);
    unsigned short* kb    = (unsigned short*)alloc(8ull * 12 * 512 * 64 * 2);
    unsigned short* vb    = (unsigned short*)alloc(8ull * 12 * 512 * 64 * 2);
    float*          ctx   = (float*)alloc(4096ull * 768 * 4);
    float*          tmp   = (float*)alloc(4096ull * 768 * 4);
    float*          attn  = (float*)alloc(4096ull * 768 * 4);
    float*          inter = (float*)alloc(4096ull * 3072 * 4);            // ~115 MB total

    hipMemcpyAsync(h, hidden, 4096ull * 768 * 4, hipMemcpyDeviceToDevice, stream);
    build_idx<<<8192, 256, 0, stream>>>(position_ids, bbox, amask, pidx);

    for (int l = 0; l < 12; ++l) {
        gemm_bf16_wmma<0><<<dim3(18, 32), 256, 0, stream>>>(
            h, qkv_w + (size_t)l * 768 * 2304, 768, 2304,
            nullptr, nullptr, nullptr,
            q_bias + (size_t)l * 768, v_bias + (size_t)l * 768, qb, kb, vb);
        attn_flash<<<dim3(8, 12, 8), 128, 0, stream>>>(
            qb, kb, vb, pidx, rel_pos_w, rel_x_w, rel_y_w, ctx);
        gemm_bf16_wmma<1><<<dim3(6, 32), 256, 0, stream>>>(
            ctx, attn_out_w + (size_t)l * 768 * 768, 768, 768,
            attn_out_b + (size_t)l * 768, h, tmp,
            nullptr, nullptr, nullptr, nullptr, nullptr);
        layernorm_768<<<4096, 256, 0, stream>>>(
            tmp, ln1_g + (size_t)l * 768, ln1_b + (size_t)l * 768, attn);
        gemm_bf16_wmma<2><<<dim3(24, 32), 256, 0, stream>>>(
            attn, inter_w + (size_t)l * 768 * 3072, 768, 3072,
            inter_b + (size_t)l * 3072, nullptr, inter,
            nullptr, nullptr, nullptr, nullptr, nullptr);
        gemm_bf16_wmma<1><<<dim3(6, 32), 256, 0, stream>>>(
            inter, out_w + (size_t)l * 3072 * 768, 3072, 768,
            out_b + (size_t)l * 768, attn, tmp,
            nullptr, nullptr, nullptr, nullptr, nullptr);
        layernorm_768<<<4096, 256, 0, stream>>>(
            tmp, ln2_g + (size_t)l * 768, ln2_b + (size_t)l * 768, h);
    }
}
